// DDiTAttention_39376260170427
// MI455X (gfx1250) — matrix operand order repaired and checked by reading, not compiled, not run on hardware
//
#include <hip/hip_runtime.h>
#include <hip/hip_bf16.h>

// DDiT attention for MI455X (gfx1250, wave32, WMMA + TDM).
// B=2, S=2048, H=12, D=64. Outputs: x [B,H,S,D] f32 ++ attn_logits [B,H,S,S] f32.
//
// Roofline: the 403 MB logits stream dominates (~17 us @ 23.3 TB/s); dynamic
// VALU count is trimmed to stay under that floor. Single pass: logits written
// once (non-temporal), flash softmax with wave-shared max (one shuffle tree per
// 64-key tile, rescale skipped when the max is stable), row sums via a
// ones-column WMMA, K and pre-transposed V tiles DMA'd by the Tensor Data
// Mover into bank-padded LDS (all fragments load as contiguous ds_load_b128).
// The 1/sqrt(D) scale is folded into Q at rotary time.

typedef __attribute__((ext_vector_type(16))) _Float16 v16h;
typedef __attribute__((ext_vector_type(8)))  float    v8f;
typedef __attribute__((ext_vector_type(4)))  unsigned int v4u;
typedef __attribute__((ext_vector_type(8)))  int      v8i;
typedef __attribute__((ext_vector_type(4)))  int      v4i;

#define B_ 2
#define S_ 2048
#define H_ 12
#define D_ 64
#define NQK (B_*H_*S_*D_)      // 3145728 elements per tensor
#define JT  64                 // keys per tile
#define NT  (S_/JT)            // 32 tiles
#define LSTR 72                // LDS row stride in halves: 144B, 16B-aligned, bank-spread

#if __has_builtin(__builtin_amdgcn_tensor_load_to_lds) && __has_builtin(__builtin_amdgcn_s_wait_tensorcnt)
#define USE_TDM 1
#else
#define USE_TDM 0
#endif

union F16Frag {
    v16h  v;
    uint4 q[2];
    _Float16 h[16];
};

#define WMMA_F16(a, b, c) \
    __builtin_amdgcn_wmma_f32_16x16x32_f16(false, (a), false, (b), (short)0, (c), false, false)

// ---------------------------------------------------------------------------
// Kernel 1: rotary for q/k slices -> f16 [B,H,S,D]; Q pre-scaled by 1/sqrt(D).
// ---------------------------------------------------------------------------
__global__ __launch_bounds__(256) void rotary_pack_kernel(
    const float* __restrict__ qkv, const float* __restrict__ cosb,
    const float* __restrict__ sinb, _Float16* __restrict__ Qh,
    _Float16* __restrict__ Kh)
{
    int tid = blockIdx.x * blockDim.x + threadIdx.x;
    const int total = B_ * S_ * 2 * H_ * D_;
    if (tid >= total) return;
    const int d = tid & (D_ - 1);
    const int h = (tid >> 6) % H_;
    const int t = (tid / (D_ * H_)) % 2;
    const int s = (tid / (D_ * H_ * 2)) % S_;
    const int b = tid / (D_ * H_ * 2 * S_);

    const size_t qi = ((size_t)(b * S_ + s) * 3 + t) * (H_ * D_) + h * D_ + d;
    const float x  = qkv[qi];
    const float xr = (d < 32) ? -qkv[qi + 32] : qkv[qi - 32];
    const float c  = cosb[(s * 3 + t) * D_ + d];
    const float sn = sinb[(s * 3 + t) * D_ + d];
    float y = x * c + xr * sn;
    if (t == 0) y *= 0.125f;            // fold 1/sqrt(64) into Q

    _Float16* dst = (t == 0) ? Qh : Kh;
    dst[((size_t)(b * H_ + h) * S_ + s) * D_ + d] = (_Float16)y;
}

// ---------------------------------------------------------------------------
// Kernel 1b: V slice (rotary is identity there) -> transposed f16 Vt [B,H,D,S].
// ---------------------------------------------------------------------------
__global__ __launch_bounds__(256) void vtrans_kernel(
    const float* __restrict__ qkv, _Float16* __restrict__ Vt)
{
    __shared__ __align__(16) _Float16 tld[64 * LSTR];
    const int bh = blockIdx.x / (S_ / 64);
    const int s0 = (blockIdx.x % (S_ / 64)) * 64;
    const int b  = bh / H_, h = bh % H_;
    const int row  = threadIdx.x >> 2;         // 0..63
    const int part = (threadIdx.x & 3) * 16;   // 0,16,32,48
    {
        const float* src = qkv + ((size_t)(b * S_ + s0 + row) * 3 + 2) * (H_ * D_) + h * D_ + part;
#pragma unroll
        for (int i = 0; i < 16; ++i)
            tld[row * LSTR + part + i] = (_Float16)src[i];
    }
    __syncthreads();
    {
        F16Frag rg;   // phase 2: row = d, part = s-chunk
#pragma unroll
        for (int i = 0; i < 16; ++i)
            rg.h[i] = tld[(part + i) * LSTR + row];
        _Float16* dst = Vt + (size_t)bh * D_ * S_ + (size_t)row * S_ + s0 + part;
        *(uint4*)(dst)     = rg.q[0];
        *(uint4*)(dst + 8) = rg.q[1];
    }
}

// ---------------------------------------------------------------------------
// TDM: DMA one 64-row x 64-elem f16 tile into LDS, 16B pad per 128B row
// (-> 72-half LDS stride). D# per ISA §8.
// ---------------------------------------------------------------------------
#if USE_TDM
__device__ __forceinline__ void tdm_load_tile(const _Float16* gsrc, unsigned lds_off,
                                              unsigned tdim0, unsigned tdim1, unsigned stride0)
{
    const unsigned long long ga = (unsigned long long)(size_t)gsrc;
    v4u g0;
    g0[0] = 1u;                                    // count=1, user descriptor
    g0[1] = lds_off;                               // lds_addr (bytes)
    g0[2] = (unsigned)(ga & 0xffffffffu);          // global_addr[31:0]
    g0[3] = (unsigned)((ga >> 32) & 0x1ffffffu) | (2u << 30);  // ga[56:32] | type=2
    v8i g1;
    g1[0] = (int)((1u << 16)      // data_size = 2 bytes
                | (1u << 20)      // pad_enable
                | (4u << 22)      // pad_interval: 32 DWORDs (one 128B row)
                | (3u << 25));    // pad_amount: 4 DWORDs (16B)
    g1[1] = (int)(tdim0 << 16);                          // tensor_dim0 lo
    g1[2] = (int)((tdim0 >> 16) | (tdim1 << 16));        // tensor_dim0 hi | tensor_dim1 lo
    g1[3] = (int)(((tdim1 >> 16) & 0xffffu) | (64u << 16)); // tensor_dim1 hi | tile_dim0=64
    g1[4] = (int)(64u);                                  // tile_dim1=64, tile_dim2=0
    g1[5] = (int)(stride0);                              // tensor_dim0_stride lo
    g1[6] = 0;
    g1[7] = 0;
    v4i z4 = {0, 0, 0, 0};
#if __clang_major__ >= 23
    v8i z8 = {0, 0, 0, 0, 0, 0, 0, 0};
    __builtin_amdgcn_tensor_load_to_lds(g0, g1, z4, z4, z8, 0);
#else
    __builtin_amdgcn_tensor_load_to_lds(g0, g1, z4, z4, 0);
#endif
}
#endif

// ---------------------------------------------------------------------------
// Kernel 2: fused attention. 128 threads = 4 waves; wave w owns 16 query rows;
// all 4 waves share (b,h) and the double-buffered K/Vt tiles.
// ---------------------------------------------------------------------------
__global__ __launch_bounds__(128) void attn_kernel(
    const _Float16* __restrict__ Qh, const _Float16* __restrict__ Kh,
    const _Float16* __restrict__ Vt, const int* __restrict__ seqlens,
    float* __restrict__ xout, float* __restrict__ lout)
{
    __shared__ __align__(16) _Float16 Klds[2][JT * LSTR];   // K  tile: [key][d]
    __shared__ __align__(16) _Float16 Vlds[2][JT * LSTR];   // Vt tile: [d][key]
    __shared__ __align__(16) _Float16 Plds[4][16 * LSTR];

    const int lane  = threadIdx.x & 31;
    const int wid   = threadIdx.x >> 5;
    const int gw    = blockIdx.x * 4 + wid;
    const int ib    = gw & 127;
    const int h     = (gw >> 7) % H_;
    const int b     = gw / (128 * H_);
    const int bh    = b * H_ + h;
    const int i0    = ib * 16;
    const int sl    = seqlens[b];

    const int n     = lane & 15;       // A-frag row / B-frag col / C-frag N
    const int hb8   = (lane >> 4) * 8; // C-frag row base
    const int halfk = lane >> 4;       // B-frag K half

    const _Float16* qrow = Qh + ((size_t)bh * S_ + i0 + n) * D_;
    F16Frag qa[2];
#pragma unroll
    for (int kb = 0; kb < 2; ++kb) {
        qa[kb].q[0] = *(const uint4*)(qrow + kb * 32 + halfk * 16);
        qa[kb].q[1] = *(const uint4*)(qrow + kb * 32 + halfk * 16 + 8);
    }

    F16Frag ones;   // o4 += P * 1 : row sums = softmax denominator
#pragma unroll
    for (int i = 0; i < 16; ++i) ones.h[i] = (_Float16)1.0f;

    v8f oA[4] = {{}, {}, {}, {}};
    v8f o4 = {};
    float Mrun = -3e38f;

    float* Lrow = lout + (size_t)bh * S_ * S_;
    const _Float16* gKbase = Kh + (size_t)bh * S_ * D_;
    const _Float16* gVbase = Vt + (size_t)bh * D_ * S_;

#if USE_TDM
    if (wid == 0) {   // wave-level DMA issue (EXEC-independent, TENSORcnt-tracked)
        tdm_load_tile(gKbase, (unsigned)(size_t)&Klds[0][0], 64u,   2048u, 64u);
        tdm_load_tile(gVbase, (unsigned)(size_t)&Vlds[0][0], 2048u, 64u,   2048u);
    }
#endif

    for (int jt = 0; jt < NT; ++jt) {
        const int j0 = jt * JT;
        int buf;
#if USE_TDM
        if (wid == 0) __builtin_amdgcn_s_wait_tensorcnt(0);
        __syncthreads();            // tile jt visible to all waves
        if (wid == 0 && jt + 1 < NT) {
            tdm_load_tile(gKbase + (size_t)(j0 + JT) * D_, (unsigned)(size_t)&Klds[(jt + 1) & 1][0],
                          64u, 2048u, 64u);
            tdm_load_tile(gVbase + (size_t)(j0 + JT),      (unsigned)(size_t)&Vlds[(jt + 1) & 1][0],
                          2048u, 64u, 2048u);
        }
        buf = jt & 1;
#else
        __syncthreads();
        {
            const int row  = threadIdx.x >> 1;
            const int part = (threadIdx.x & 1) * 32;
            const _Float16* gk = gKbase + (size_t)(j0 + row) * D_ + part;
            const _Float16* gv = gVbase + (size_t)row * S_ + j0 + part;
            _Float16* dk = &Klds[0][row * LSTR + part];
            _Float16* dv = &Vlds[0][row * LSTR + part];
#pragma unroll
            for (int u = 0; u < 4; ++u) {
                *(uint4*)(dk + u * 8) = *(const uint4*)(gk + u * 8);
                *(uint4*)(dv + u * 8) = *(const uint4*)(gv + u * 8);
            }
            if (jt + 1 < NT) {
                __builtin_prefetch(gk + JT * D_, 0, 1);
                __builtin_prefetch(gv + JT, 0, 1);
            }
        }
        __syncthreads();
        buf = 0;
#endif
        const _Float16* Kb = &Klds[buf][0];
        const _Float16* Vb = &Vlds[buf][0];

        // ---- logits (pre-scaled via Q): 4 N-subtiles x 2 K-steps = 8 WMMA ----
        v8f s0 = {}, s1 = {}, s2 = {}, s3 = {};
#pragma unroll
        for (int kb = 0; kb < 2; ++kb) {
            const int ko = kb * 32 + halfk * 16;
            F16Frag bf;
            bf.v = *(const v16h*)&Kb[(n)      * LSTR + ko];
            s0 = WMMA_F16(qa[kb].v, bf.v, s0);
            bf.v = *(const v16h*)&Kb[(16 + n) * LSTR + ko];
            s1 = WMMA_F16(qa[kb].v, bf.v, s1);
            bf.v = *(const v16h*)&Kb[(32 + n) * LSTR + ko];
            s2 = WMMA_F16(qa[kb].v, bf.v, s2);
            bf.v = *(const v16h*)&Kb[(48 + n) * LSTR + ko];
            s3 = WMMA_F16(qa[kb].v, bf.v, s3);
        }

        // ---- mask (only on boundary tiles) + stream logits + tile max ----
        const bool fullTile = ((j0 + JT) <= sl) && ((i0 + 16) <= sl);
        float tAll = -1e10f;
        if (fullTile) {
#pragma unroll
            for (int r = 0; r < 8; ++r) {
                const int gi = i0 + r + hb8;
                float* lp = Lrow + (size_t)gi * S_ + j0 + n;
                __builtin_nontemporal_store(s0[r], lp);
                __builtin_nontemporal_store(s1[r], lp + 16);
                __builtin_nontemporal_store(s2[r], lp + 32);
                __builtin_nontemporal_store(s3[r], lp + 48);
                tAll = fmaxf(tAll, fmaxf(fmaxf(s0[r], s1[r]), fmaxf(s2[r], s3[r])));
            }
        } else {
            const bool cv0 = (j0 + n)      < sl;
            const bool cv1 = (j0 + 16 + n) < sl;
            const bool cv2 = (j0 + 32 + n) < sl;
            const bool cv3 = (j0 + 48 + n) < sl;
#pragma unroll
            for (int r = 0; r < 8; ++r) {
                const int  gi = i0 + r + hb8;
                const bool rv = gi < sl;
                const float v0 = (rv && cv0) ? s0[r] : -1e10f;
                const float v1 = (rv && cv1) ? s1[r] : -1e10f;
                const float v2 = (rv && cv2) ? s2[r] : -1e10f;
                const float v3 = (rv && cv3) ? s3[r] : -1e10f;
                s0[r] = v0; s1[r] = v1; s2[r] = v2; s3[r] = v3;   // reuse in P pass
                float* lp = Lrow + (size_t)gi * S_ + j0 + n;
                __builtin_nontemporal_store(v0, lp);
                __builtin_nontemporal_store(v1, lp + 16);
                __builtin_nontemporal_store(v2, lp + 32);
                __builtin_nontemporal_store(v3, lp + 48);
                tAll = fmaxf(tAll, fmaxf(fmaxf(v0, v1), fmaxf(v2, v3)));
            }
        }
#pragma unroll
        for (int msk = 16; msk >= 1; msk >>= 1)
            tAll = fmaxf(tAll, __shfl_xor(tAll, msk, 32));

        if (tAll > Mrun) {   // wave-uniform: rescale only when the max grows
            const float corr = __expf(Mrun - tAll);
            Mrun = tAll;
#pragma unroll
            for (int r = 0; r < 8; ++r) {   // dead rows: corr=1 -> uniform softmax
                const float cr = ((i0 + r + hb8) < sl) ? corr : 1.0f;
                oA[0][r] *= cr; oA[1][r] *= cr; oA[2][r] *= cr; oA[3][r] *= cr; o4[r] *= cr;
            }
        }

        // ---- P = exp(s - Mrun) -> per-wave LDS (f16) ----
        _Float16* Pw = &Plds[wid][0];
        if (fullTile) {
#pragma unroll
            for (int r = 0; r < 8; ++r) {
                const int M = r + hb8;
                Pw[M * LSTR + n]      = (_Float16)__expf(s0[r] - Mrun);
                Pw[M * LSTR + 16 + n] = (_Float16)__expf(s1[r] - Mrun);
                Pw[M * LSTR + 32 + n] = (_Float16)__expf(s2[r] - Mrun);
                Pw[M * LSTR + 48 + n] = (_Float16)__expf(s3[r] - Mrun);
            }
        } else {
#pragma unroll
            for (int r = 0; r < 8; ++r) {
                const int  M  = r + hb8;
                const bool rv = (i0 + M) < sl;
                const float p0 = rv ? __expf(s0[r] - Mrun) : 1.0f;  // masked cols -> 0
                const float p1 = rv ? __expf(s1[r] - Mrun) : 1.0f;  // dead rows -> 1
                const float p2 = rv ? __expf(s2[r] - Mrun) : 1.0f;
                const float p3 = rv ? __expf(s3[r] - Mrun) : 1.0f;
                Pw[M * LSTR + n]      = (_Float16)p0;
                Pw[M * LSTR + 16 + n] = (_Float16)p1;
                Pw[M * LSTR + 32 + n] = (_Float16)p2;
                Pw[M * LSTR + 48 + n] = (_Float16)p3;
            }
        }
        __syncthreads();

        // ---- O += P V  and  o4 += P 1 : 10 WMMA, all-contiguous LDS loads ----
        F16Frag pa0, pa1;
        pa0.q[0] = *(const uint4*)&Pw[n * LSTR + halfk * 16];
        pa0.q[1] = *(const uint4*)&Pw[n * LSTR + halfk * 16 + 8];
        pa1.q[0] = *(const uint4*)&Pw[n * LSTR + 32 + halfk * 16];
        pa1.q[1] = *(const uint4*)&Pw[n * LSTR + 32 + halfk * 16 + 8];

        o4 = WMMA_F16(pa0.v, ones.v, o4);
        o4 = WMMA_F16(pa1.v, ones.v, o4);
#pragma unroll
        for (int kk = 0; kk < 2; ++kk) {
            const v16h pav = kk ? pa1.v : pa0.v;
#pragma unroll
            for (int c = 0; c < 4; ++c) {
                F16Frag vb;   // B(k,n) = Vt[c*16+n][j0 + kk*32 + k]: contiguous 32B
                vb.v = *(const v16h*)&Vb[(c * 16 + n) * LSTR + kk * 32 + halfk * 16];
                oA[c] = WMMA_F16(pav, vb.v, oA[c]);
            }
        }
    }

    // ---- finalize: x = O / l ----
#pragma unroll
    for (int r = 0; r < 8; ++r) {
        const int M = r + hb8;
        const float inv = 1.0f / o4[r];
        float* xp = xout + ((size_t)bh * S_ + i0 + M) * D_ + n;
        __builtin_nontemporal_store(oA[0][r] * inv, xp);
        __builtin_nontemporal_store(oA[1][r] * inv, xp + 16);
        __builtin_nontemporal_store(oA[2][r] * inv, xp + 32);
        __builtin_nontemporal_store(oA[3][r] * inv, xp + 48);
    }
}

// ---------------------------------------------------------------------------
extern "C" void kernel_launch(void* const* d_in, const int* in_sizes, int n_in,
                              void* d_out, int out_size, void* d_ws, size_t ws_size,
                              hipStream_t stream) {
    const float* qkv     = (const float*)d_in[0];   // [B,S,3,H,D]
    const float* cosb    = (const float*)d_in[1];   // [1,S,3,1,D]
    const float* sinb    = (const float*)d_in[2];   // [1,S,3,1,D]
    const int*   seqlens = (const int*)d_in[3];     // [B]

    _Float16* Qh = (_Float16*)d_ws;                 // 3 x 6.29 MB f16 workspace
    _Float16* Kh = Qh + NQK;
    _Float16* Vt = Kh + NQK;                        // transposed V: [B,H,D,S]

    float* xout = (float*)d_out;                    // [B,H,S,D]
    float* lout = xout + (size_t)B_ * H_ * S_ * D_; // [B,H,S,S]

    {
        const int total = B_ * S_ * 2 * H_ * D_;    // q/k slices only
        rotary_pack_kernel<<<(total + 255) / 256, 256, 0, stream>>>(
            qkv, cosb, sinb, Qh, Kh);
    }
    {
        const int nblk = B_ * H_ * (S_ / 64);       // 768 transpose tiles
        vtrans_kernel<<<nblk, 256, 0, stream>>>(qkv, Vt);
    }
    {
        const int nwaves = B_ * H_ * (S_ / 16);     // 3072 waves, 4 per block
        attn_kernel<<<nwaves / 4, 128, 0, stream>>>(
            Qh, Kh, Vt, seqlens, xout, lout);
    }
}
